// STDI_Net_73091753443564
// MI455X (gfx1250) — compile-verified, matrix operand order checked by reading, not run on hardware
//
#include <hip/hip_runtime.h>
#include <hip/hip_bf16.h>
#include <math.h>
#include <stdint.h>

// ---------------------------------------------------------------------------
// STDI-Net forward for MI455X (gfx1250, wave32, WMMA).
// Heavy GEMMs (LSTM input/hidden projections, DI-net hypernetwork collapsed
// algebraically) run on v_wmma_f32_16x16x32_f16 with the shared B tile staged
// into LDS via double-buffered global_load_async_to_lds_b128 (ASYNCcnt).
// Convs/BN are exact fp32.
// ---------------------------------------------------------------------------

typedef _Float16 half_t;
typedef __attribute__((ext_vector_type(16))) _Float16 v16h;
typedef __attribute__((ext_vector_type(8)))  _Float16 v8h;
typedef __attribute__((ext_vector_type(8)))  float    v8f;
typedef __attribute__((address_space(3)))    half_t*  lds_half_ptr;

#define HGR   16
#define WGR   8
#define PIX   128        // 16*8
#define CCH   32
#define SEQT  4
#define BB    256
#define HID   1024
#define LSTM_IN 4096
#define OUTN  256
#define HOURN 24
#define BN_EPS 1e-5f

// LDS B-tile geometry: 64 rows (N) x 32 halfs (K), padded to 40 halfs (80B,
// 16B-aligned) per row to spread banks; double buffered.
#define BROW_H 40
#define BBUF_H (64 * BROW_H)           // halfs per buffer
#define BBUF_BYTES (BBUF_H * 2)

// ---------------- WMMA GEMM:  C[M,N] = A[M,K] * W[N,K]^T (+epilogue) --------
// Block = 128 threads = 4 waves; block tile 64x64; wave tile 16x64.
// f16 fragment layout (ISA 7.12.2, 16-bit A 16x32): lane half L<16 holds row
// M=L with K = {t*8..t*8+7, 16+t*8..16+t*8+7} (t = lane>>4); B 32x16 mirrors
// with lane giving N. Each fragment is two contiguous 16-byte runs per lane.
__device__ inline v16h load_frag16(const half_t* __restrict__ base, int row,
                                   int ldk, int k0, int lane) {
    const int t = lane >> 4;
    const half_t* p = base + (size_t)row * ldk + k0 + t * 8;
    union { v16h v; v8h h[2]; } u;
    u.h[0] = *(const v8h*)(p);
    u.h[1] = *(const v8h*)(p + 16);
    return u.v;
}

// Issue the 64x32-half B tile as 256 async 16-byte LDS fills (2 per thread).
__device__ inline void prefetch_B_async(const half_t* __restrict__ W, int n0,
                                        int K, int k0, unsigned ldsBufByte,
                                        int tid) {
    #pragma unroll
    for (int s = 0; s < 2; ++s) {
        const int L = tid + s * 128;      // 0..255
        const int row = L >> 2;           // 0..63
        const int q = L & 3;              // 16B quarter within the 64B row
        const half_t* g = W + (size_t)(n0 + row) * K + k0 + q * 8;
        const unsigned l = ldsBufByte + row * (BROW_H * 2) + q * 16;
        asm volatile("global_load_async_to_lds_b128 %0, %1, off"
                     :: "v"(l), "v"((unsigned long long)(uintptr_t)g)
                     : "memory");
    }
}

__global__ __launch_bounds__(128)
void gemm_wmma_f16(const half_t* __restrict__ A, const half_t* __restrict__ W,
                   float* __restrict__ C, int M, int N, int K,
                   const float* __restrict__ addmat,
                   const float* __restrict__ biasvec, int relu) {
    __shared__ half_t ldsB[2 * BBUF_H];   // double-buffered B tile (~10KB)
    const int tid  = threadIdx.x;
    const int lane = tid & 31;
    const int wave = tid >> 5;
    const int m0 = blockIdx.x * 64 + wave * 16;
    const int n0 = blockIdx.y * 64;
    const int rA = m0 + (lane & 15);
    const int t = lane >> 4;
    const int nlo = lane & 15;
    const unsigned ldsBase = (unsigned)(uintptr_t)(lds_half_ptr)ldsB;

    v8f acc[4];
    for (int j = 0; j < 4; ++j)
        for (int i = 0; i < 8; ++i) acc[j][i] = 0.f;

    const int steps = K >> 5;
    prefetch_B_async(W, n0, K, 0, ldsBase, tid);               // fill buf 0

    for (int i = 0; i < steps; ++i) {
        const int k0 = i << 5;
        const int cur = i & 1;
        // Only the current tile's fill is outstanding here.
        asm volatile("s_wait_asynccnt 0x0" ::: "memory");
        __syncthreads();   // cur tile visible; all waves done reading cur^1
        if (i + 1 < steps)
            prefetch_B_async(W, n0, K, k0 + 32,
                             ldsBase + (cur ^ 1) * BBUF_BYTES, tid);

        v16h a = load_frag16(A, rA, K, k0, lane);
        const half_t* bbase = ldsB + cur * BBUF_H;
        #pragma unroll
        for (int j = 0; j < 4; ++j) {
            const half_t* rowp = bbase + (j * 16 + nlo) * BROW_H;
            union { v16h v; v8h h[2]; } u;
            u.h[0] = *(const v8h*)(rowp + t * 8);
            u.h[1] = *(const v8h*)(rowp + 16 + t * 8);
            acc[j] = __builtin_amdgcn_wmma_f32_16x16x32_f16(
                false, a, false, u.v, (short)0, acc[j], false, false);
        }
    }
    // C/D layout: VGPR r, lane L -> n = L%16, m = r + 8*(L>>4)
    for (int j = 0; j < 4; ++j) {
        int n = n0 + j * 16 + nlo;
        #pragma unroll
        for (int r = 0; r < 8; ++r) {
            int m = m0 + r + 8 * t;
            size_t idx = (size_t)m * N + n;
            float v = acc[j][r];
            if (addmat)  v += addmat[idx];
            if (biasvec) v += biasvec[n];
            if (relu)    v = fmaxf(v, 0.f);
            C[idx] = v;
        }
    }
}

// ---------------- direct 3x3 'SAME' conv, fp32, LDS-staged -----------------
__global__ __launch_bounds__(128)
void conv3x3_kernel(const float* __restrict__ xin, int Cin,
                    const float* __restrict__ w, const float* __restrict__ bias,
                    float* __restrict__ y) {
    extern __shared__ float smem[];
    float* xs = smem;               // Cin*PIX
    float* ws = smem + Cin * PIX;   // CCH*Cin*9
    const int b = blockIdx.x;
    const int tid = threadIdx.x;
    for (int i = tid; i < Cin * PIX; i += 128)
        xs[i] = xin[(size_t)b * Cin * PIX + i];
    for (int i = tid; i < CCH * Cin * 9; i += 128)
        ws[i] = w[i];
    __syncthreads();
    const int p = tid, py = p >> 3, px = p & 7;
    for (int co = 0; co < CCH; ++co) {
        float acc = bias[co];
        const float* wc = ws + co * Cin * 9;
        for (int ci = 0; ci < Cin; ++ci) {
            const float* xc = xs + ci * PIX;
            const float* wk = wc + ci * 9;
            #pragma unroll
            for (int ky = 0; ky < 3; ++ky) {
                int yy = py + ky - 1;
                if (yy < 0 || yy >= HGR) continue;
                #pragma unroll
                for (int kx = 0; kx < 3; ++kx) {
                    int xx = px + kx - 1;
                    if (xx < 0 || xx >= WGR) continue;
                    acc = fmaf(wk[ky * 3 + kx], xc[yy * 8 + xx], acc);
                }
            }
        }
        y[((size_t)b * CCH + co) * PIX + p] = acc;
    }
}

// ---------------- BatchNorm (training stats, biased var) -------------------
__global__ __launch_bounds__(256)
void bn_stats_kernel(const float* __restrict__ y, const float* __restrict__ g,
                     const float* __restrict__ be, float* __restrict__ scale,
                     float* __restrict__ shift) {
    __shared__ float ssum[256], ssq[256];
    const int c = blockIdx.x, tid = threadIdx.x;
    float s = 0.f, q = 0.f;
    for (int i = tid; i < BB * PIX; i += 256) {
        int b = i >> 7, p = i & 127;
        float v = y[((size_t)b * CCH + c) * PIX + p];
        s += v; q += v * v;
    }
    ssum[tid] = s; ssq[tid] = q;
    __syncthreads();
    for (int off = 128; off > 0; off >>= 1) {
        if (tid < off) { ssum[tid] += ssum[tid + off]; ssq[tid] += ssq[tid + off]; }
        __syncthreads();
    }
    if (tid == 0) {
        const float invN = 1.f / (float)(BB * PIX);
        float mean = ssum[0] * invN;
        float var  = ssq[0] * invN - mean * mean;
        float inv  = rsqrtf(var + BN_EPS);
        float sc   = g[c] * inv;
        scale[c] = sc;
        shift[c] = be[c] - mean * sc;
    }
}

__global__ void bn_apply_kernel(const float* a, const float* scale,
                                const float* shift, float* o) {
    int i = blockIdx.x * blockDim.x + threadIdx.x;
    if (i >= BB * CCH * PIX) return;
    int c = (i >> 7) & 31;
    o[i] = a[i] * scale[c] + shift[c];
}

__global__ void bn_apply_add_relu_kernel(const float* a, const float* res,
                                         const float* scale, const float* shift,
                                         float* o) {
    int i = blockIdx.x * blockDim.x + threadIdx.x;
    if (i >= BB * CCH * PIX) return;
    int c = (i >> 7) & 31;
    o[i] = fmaxf(res[i] + a[i] * scale[c] + shift[c], 0.f);
}

__global__ void bn_apply_relu_f16_kernel(const float* a, const float* scale,
                                         const float* shift, half_t* feats) {
    int i = blockIdx.x * blockDim.x + threadIdx.x;
    if (i >= BB * CCH * PIX) return;
    int c = (i >> 7) & 31;
    feats[i] = (half_t)fmaxf(a[i] * scale[c] + shift[c], 0.f);
}

// ---------------- small helpers -------------------------------------------
__global__ void cast_f16_kernel(const float* src, half_t* dst, int n) {
    int i = blockIdx.x * blockDim.x + threadIdx.x;
    if (i < n) dst[i] = (half_t)src[i];
}
__global__ void bias_sum_kernel(const float* a, const float* b, float* o, int n) {
    int i = blockIdx.x * blockDim.x + threadIdx.x;
    if (i < n) o[i] = a[i] + b[i];
}
__global__ void zero_f16_kernel(half_t* p, int n) {
    int i = blockIdx.x * blockDim.x + threadIdx.x;
    if (i < n) p[i] = (half_t)0.f;
}
__global__ void dinet_fc_kernel(const float* __restrict__ h,
                                const float* __restrict__ Wt,
                                const float* __restrict__ bias,
                                float* __restrict__ out, int Nout) {
    int i = blockIdx.x * blockDim.x + threadIdx.x;
    if (i >= BB * Nout) return;
    int b = i / Nout, n = i % Nout;
    float acc = bias[n];
    const float* hb = h + b * HOURN;
    const float* wn = Wt + n * HOURN;
    #pragma unroll
    for (int k = 0; k < HOURN; ++k) acc = fmaf(hb[k], wn[k], acc);
    out[i] = acc > 0.f ? acc : 0.01f * acc;
}

__global__ void lstm_gate_kernel(const float* __restrict__ z, float* h32,
                                 half_t* h16, float* c, int first) {
    int i = blockIdx.x * blockDim.x + threadIdx.x;
    if (i >= BB * HID) return;
    int b = i >> 10, u = i & 1023;
    const float* zb = z + (size_t)b * 4 * HID;
    float zi = zb[u], zf = zb[HID + u], zg = zb[2 * HID + u], zo = zb[3 * HID + u];
    float si = 1.f / (1.f + expf(-zi));
    float sf = 1.f / (1.f + expf(-zf));
    float so = 1.f / (1.f + expf(-zo));
    float cold = first ? 0.f : c[i];
    float cn = sf * cold + si * tanhf(zg);
    float hn = so * tanhf(cn);
    c[i] = cn; h32[i] = hn; h16[i] = (half_t)hn;
}

__global__ void scale_u_kernel(const float* __restrict__ wres,
                               const float* __restrict__ S,
                               half_t* __restrict__ U) {
    int i = blockIdx.x * blockDim.x + threadIdx.x;
    if (i >= BB * 2048) return;
    int b = i >> 11, rj = i & 2047, r = rj >> 9, j = rj & 511;
    U[i] = (half_t)(wres[b * 512 + j] * S[(size_t)(b * 4 + r) * 512 + j]);
}

// ---------------------------------------------------------------------------
extern "C" void kernel_launch(void* const* d_in, const int* in_sizes, int n_in,
                              void* d_out, int out_size, void* d_ws, size_t ws_size,
                              hipStream_t stream) {
    (void)in_sizes; (void)n_in; (void)out_size; (void)ws_size;
    const float* x   = (const float*)d_in[0];   // [4,256,2,16,8]
    const float* h24 = (const float*)d_in[1];   // [256,24]
    auto SP = [&](int t, int k) { return (const float*)d_in[2 + t * 20 + k]; };
    const float* Wih = (const float*)d_in[82];  // [4096,4096]
    const float* Whh = (const float*)d_in[83];  // [4096,1024]
    const float* bih = (const float*)d_in[84];
    const float* bhh = (const float*)d_in[85];
    const float* Wm  = (const float*)d_in[86];  // [512,24]
    const float* bm  = (const float*)d_in[87];
    const float* Wb  = (const float*)d_in[88];  // [256,24]
    const float* bbv = (const float*)d_in[89];
    const float* left  = (const float*)d_in[90]; // [1024,512] == view [256,2048]
    const float* right = (const float*)d_in[91]; // [512,256]
    float* out = (float*)d_out;                  // [256,256]

    char* cur = (char*)d_ws;
    auto carve = [&](size_t bytes) -> void* {
        void* p = cur; cur += (bytes + 255) & ~(size_t)255; return p;
    };
    half_t* Wih16   = (half_t*)carve((size_t)4096 * 4096 * 2);
    half_t* Whh16   = (half_t*)carve((size_t)4096 * 1024 * 2);
    half_t* right16 = (half_t*)carve((size_t)512 * 256 * 2);
    half_t* left16  = (half_t*)carve((size_t)1024 * 512 * 2);
    float* bsum   = (float*)carve(4096 * 4);
    float* wres   = (float*)carve((size_t)BB * 512 * 4);
    float* dibias = (float*)carve((size_t)BB * 256 * 4);
    float* bufA = (float*)carve((size_t)BB * CCH * PIX * 4);
    float* bufB = (float*)carve((size_t)BB * CCH * PIX * 4);
    float* bufO = (float*)carve((size_t)BB * CCH * PIX * 4);
    float* bufY = (float*)carve((size_t)BB * CCH * PIX * 4);
    float* scb  = (float*)carve(CCH * 4);
    float* shb  = (float*)carve(CCH * 4);
    half_t* feats16 = (half_t*)carve((size_t)SEQT * BB * LSTM_IN * 2);
    float* xz   = (float*)carve((size_t)SEQT * BB * 4096 * 4);
    float* zbuf = (float*)carve((size_t)BB * 4096 * 4);
    float* h32  = (float*)carve((size_t)BB * HID * 4);
    half_t* h16 = (half_t*)carve((size_t)BB * HID * 2);
    float* cbuf = (float*)carve((size_t)BB * HID * 4);
    float* Sbuf = (float*)carve((size_t)BB * 4 * 512 * 4);
    half_t* U16 = (half_t*)carve((size_t)BB * 2048 * 2);

    const int TPB = 256;
    auto g1 = [&](int n) { return dim3((unsigned)((n + TPB - 1) / TPB)); };

    // weight precision conversion + fused LSTM bias
    cast_f16_kernel<<<g1(4096 * 4096), TPB, 0, stream>>>(Wih, Wih16, 4096 * 4096);
    cast_f16_kernel<<<g1(4096 * 1024), TPB, 0, stream>>>(Whh, Whh16, 4096 * 1024);
    cast_f16_kernel<<<g1(512 * 256),  TPB, 0, stream>>>(right, right16, 512 * 256);
    cast_f16_kernel<<<g1(1024 * 512), TPB, 0, stream>>>(left, left16, 1024 * 512);
    bias_sum_kernel<<<g1(4096), TPB, 0, stream>>>(bih, bhh, bsum, 4096);

    // DI-net small FCs (K=24)
    dinet_fc_kernel<<<g1(BB * 512), TPB, 0, stream>>>(h24, Wm, bm, wres, 512);
    dinet_fc_kernel<<<g1(BB * 256), TPB, 0, stream>>>(h24, Wb, bbv, dibias, 256);

    // per-timestep conv blocks (exact fp32, batch-stat BN)
    const int NE = BB * CCH * PIX;
    const size_t shm2  = (size_t)(2 * PIX + CCH * 2 * 9) * 4;
    const size_t shm32 = (size_t)(32 * PIX + CCH * 32 * 9) * 4;
    for (int t = 0; t < SEQT; ++t) {
        const float* xt = x + (size_t)t * BB * 2 * PIX;
        conv3x3_kernel<<<BB, 128, shm2, stream>>>(xt, 2, SP(t, 0), SP(t, 1), bufA);
        bn_stats_kernel<<<CCH, 256, 0, stream>>>(bufA, SP(t, 2), SP(t, 3), scb, shb);
        bn_apply_kernel<<<g1(NE), TPB, 0, stream>>>(bufA, scb, shb, bufO);
        conv3x3_kernel<<<BB, 128, shm32, stream>>>(bufO, 32, SP(t, 4), SP(t, 5), bufB);
        bn_stats_kernel<<<CCH, 256, 0, stream>>>(bufB, SP(t, 6), SP(t, 7), scb, shb);
        bn_apply_add_relu_kernel<<<g1(NE), TPB, 0, stream>>>(bufB, bufO, scb, shb, bufY);
        conv3x3_kernel<<<BB, 128, shm32, stream>>>(bufY, 32, SP(t, 8), SP(t, 9), bufA);
        bn_stats_kernel<<<CCH, 256, 0, stream>>>(bufA, SP(t, 10), SP(t, 11), scb, shb);
        bn_apply_kernel<<<g1(NE), TPB, 0, stream>>>(bufA, scb, shb, bufO);
        conv3x3_kernel<<<BB, 128, shm32, stream>>>(bufO, 32, SP(t, 12), SP(t, 13), bufB);
        bn_stats_kernel<<<CCH, 256, 0, stream>>>(bufB, SP(t, 14), SP(t, 15), scb, shb);
        bn_apply_add_relu_kernel<<<g1(NE), TPB, 0, stream>>>(bufB, bufO, scb, shb, bufY);
        conv3x3_kernel<<<BB, 128, shm32, stream>>>(bufY, 32, SP(t, 16), SP(t, 17), bufA);
        bn_stats_kernel<<<CCH, 256, 0, stream>>>(bufA, SP(t, 18), SP(t, 19), scb, shb);
        bn_apply_relu_f16_kernel<<<g1(NE), TPB, 0, stream>>>(
            bufA, scb, shb, feats16 + (size_t)t * BB * LSTM_IN);
    }

    // xz[T*B,4096] = feats @ Wih^T + (bih+bhh)     (WMMA, K=4096)
    gemm_wmma_f16<<<dim3(SEQT * BB / 64, 4096 / 64), 128, 0, stream>>>(
        feats16, Wih16, xz, SEQT * BB, 4096, LSTM_IN, nullptr, bsum, 0);

    // LSTM recurrence: z = xz[t] + h @ Whh^T       (WMMA, K=1024)
    zero_f16_kernel<<<g1(BB * HID), TPB, 0, stream>>>(h16, BB * HID);
    for (int t = 0; t < SEQT; ++t) {
        gemm_wmma_f16<<<dim3(BB / 64, 4096 / 64), 128, 0, stream>>>(
            h16, Whh16, zbuf, BB, 4096, HID, xz + (size_t)t * BB * 4096, nullptr, 0);
        lstm_gate_kernel<<<g1(BB * HID), TPB, 0, stream>>>(zbuf, h32, h16, cbuf,
                                                           t == 0 ? 1 : 0);
    }

    // DI-net collapsed einsum:
    // S[(b*4+r),j] = sum_m lstm[b, r*256+m] * right[j,m]  (h16 viewed [1024,256])
    gemm_wmma_f16<<<dim3(1024 / 64, 512 / 64), 128, 0, stream>>>(
        h16, right16, Sbuf, 1024, 512, 256, nullptr, nullptr, 0);
    scale_u_kernel<<<g1(BB * 2048), TPB, 0, stream>>>(wres, Sbuf, U16);
    // out[b,n] = relu( U[b,:] . left_view[n,:] + dibias[b,n] ),
    // left [1024,512] reinterpreted as [256,2048] with zero data movement.
    gemm_wmma_f16<<<dim3(BB / 64, OUTN / 64), 128, 0, stream>>>(
        U16, left16, out, BB, OUTN, 2048, dibias, nullptr, 1);
}
